// SAGENet_28707561406529
// MI455X (gfx1250) — matrix operand order repaired and checked by reading, not compiled, not run on hardware
//
#include <hip/hip_runtime.h>
#include <hip/hip_bf16.h>
#include <math.h>

// ---------------------------------------------------------------------------
// SAGENet on MI455X (gfx1250, wave32).
//  - Heavy GEMMs (~490 GFLOP) -> v_wmma_f32_16x16x32_f16, f32 accumulate.
//  - 128x128 block tile, 8 waves, each wave = 32x64 slab = 8 WMMAs/K-chunk.
//  - LDS layouts chosen so every fragment is two contiguous 16B runs per lane
//    (A row-major, B K-transposed; 112B row stride = conflict-friendly).
//  - Edge scatter (~17 GB irregular) is the true roofline bound -> L2 atomics.
// ---------------------------------------------------------------------------

typedef __attribute__((ext_vector_type(16))) _Float16 v16h;
typedef __attribute__((ext_vector_type(8)))  float    v8f;

union Frag { v16h v; float4 q[2]; _Float16 h[16]; };
union V8F  { v8f  v; float f[8]; };
union H2U  { _Float16 h[2]; unsigned int u; };

#define NN    100000
#define NB    256
#define F0    334
#define F1    334
#define F2    668
#define F3    1336
#define LMUT  13134

#define BM 128
#define BN 128
#define TK 32
#define LDSTRIDE 56   // halfs per row: 112B, multiple of 16B, 16-way-clean banks

// ---------------------------------------------------------------------------
// C[M,N] = act( A*rowScale @ B  [+ A2 @ B2]  + bias ),  all f32 row-major.
// act: 0=none, 1=relu.
// ---------------------------------------------------------------------------
template<bool DUAL, bool SCALE>
__global__ __launch_bounds__(256)
void gemm_kernel(const float* __restrict__ A,  const float* __restrict__ B,
                 const float* __restrict__ A2, const float* __restrict__ B2,
                 const float* __restrict__ scl,
                 const float* __restrict__ bias,
                 float* __restrict__ C,
                 int M, int N, int K, int act)
{
    __shared__ __align__(16) _Float16 As [BM][LDSTRIDE];   // [row][k]
    __shared__ __align__(16) _Float16 BsT[BN][LDSTRIDE];   // [col][k] (transposed)

    const int tid  = threadIdx.x;
    const int lane = tid & 31;
    const int wave = tid >> 5;
    const int wm   = wave & 3;    // 4 M-groups of 32 rows
    const int wn   = wave >> 2;   // 2 N-groups of 64 cols
    const int mrow = lane & 15;
    const int half = lane >> 4;
    const int ncol = lane & 15;

    const int rowBase = blockIdx.y * BM;
    const int colBase = blockIdx.x * BN;

    v8f acc[2][4];
    #pragma unroll
    for (int mi = 0; mi < 2; ++mi)
        #pragma unroll
        for (int ni = 0; ni < 4; ++ni) acc[mi][ni] = {};

    const int npair = DUAL ? 2 : 1;
    for (int pair = 0; pair < npair; ++pair) {
        const float* Ap = (DUAL && pair) ? A2 : A;
        const float* Bp = (DUAL && pair) ? B2 : B;
        const bool scaleThis = SCALE && (pair == 0);

        for (int k0 = 0; k0 < K; k0 += TK) {
            __syncthreads();
            // ---------------- stage A tile (BM x TK) as packed half2 ------
            const bool fullA = (rowBase + BM <= M) && (k0 + TK <= K);
            if (fullA) {
                #pragma unroll
                for (int i = 0; i < (BM * TK / 2) / 256; ++i) {
                    int idx2 = tid + i * 256;           // pair index
                    int r  = idx2 >> 4;                 // 16 pairs per row
                    int c2 = idx2 & 15;
                    int gr = rowBase + r;
                    const float* ap = Ap + (size_t)gr * K + (k0 + 2 * c2);
                    float x0 = ap[0], x1 = ap[1];
                    if (scaleThis) { float s = scl[gr]; x0 *= s; x1 *= s; }
                    H2U p; p.h[0] = (_Float16)x0; p.h[1] = (_Float16)x1;
                    *(unsigned int*)&As[r][2 * c2] = p.u;
                }
            } else {
                #pragma unroll
                for (int i = 0; i < (BM * TK / 2) / 256; ++i) {
                    int idx2 = tid + i * 256;
                    int r  = idx2 >> 4;
                    int c2 = idx2 & 15;
                    int gr = rowBase + r, gc = k0 + 2 * c2;
                    float x0 = 0.0f, x1 = 0.0f;
                    if (gr < M) {
                        float s = scaleThis ? scl[gr] : 1.0f;
                        if (gc     < K) x0 = Ap[(size_t)gr * K + gc] * s;
                        if (gc + 1 < K) x1 = Ap[(size_t)gr * K + gc + 1] * s;
                    }
                    H2U p; p.h[0] = (_Float16)x0; p.h[1] = (_Float16)x1;
                    *(unsigned int*)&As[r][2 * c2] = p.u;
                }
            }
            // ---------------- stage B tile (TK x BN) K-transposed ---------
            const bool fullB = (colBase + BN <= N) && (k0 + TK <= K);
            if (fullB) {
                #pragma unroll
                for (int i = 0; i < (BN * TK / 2) / 256; ++i) {
                    int idx2 = tid + i * 256;
                    int n  = idx2 >> 4;                 // 16 k-pairs per col
                    int k2 = idx2 & 15;
                    int gn = colBase + n;
                    const float* bp = Bp + (size_t)(k0 + 2 * k2) * N + gn;
                    float b0 = bp[0], b1 = bp[N];
                    H2U p; p.h[0] = (_Float16)b0; p.h[1] = (_Float16)b1;
                    *(unsigned int*)&BsT[n][2 * k2] = p.u;
                }
            } else {
                #pragma unroll
                for (int i = 0; i < (BN * TK / 2) / 256; ++i) {
                    int idx2 = tid + i * 256;
                    int n  = idx2 >> 4;
                    int k2 = idx2 & 15;
                    int gn = colBase + n, gk = k0 + 2 * k2;
                    float b0 = 0.0f, b1 = 0.0f;
                    if (gn < N) {
                        if (gk     < K) b0 = Bp[(size_t)gk * N + gn];
                        if (gk + 1 < K) b1 = Bp[(size_t)(gk + 1) * N + gn];
                    }
                    H2U p; p.h[0] = (_Float16)b0; p.h[1] = (_Float16)b1;
                    *(unsigned int*)&BsT[n][2 * k2] = p.u;
                }
            }
            __syncthreads();

            // ---------------- fragments: contiguous 16B runs --------------
            Frag a0, a1;
            {
                const float4* r0 = (const float4*)&As[wm * 32 + mrow][0];
                a0.q[0] = r0[half];     a0.q[1] = r0[2 + half];
                const float4* r1 = (const float4*)&As[wm * 32 + 16 + mrow][0];
                a1.q[0] = r1[half];     a1.q[1] = r1[2 + half];
            }
            Frag bf[4];
            #pragma unroll
            for (int ni = 0; ni < 4; ++ni) {
                const float4* br = (const float4*)&BsT[wn * 64 + ni * 16 + ncol][0];
                bf[ni].q[0] = br[half];
                bf[ni].q[1] = br[2 + half];
            }
            #pragma unroll
            for (int ni = 0; ni < 4; ++ni) {
                acc[0][ni] = __builtin_amdgcn_wmma_f32_16x16x32_f16(
                    false, a0.v, false, bf[ni].v, (short)0, acc[0][ni], false, false);
                acc[1][ni] = __builtin_amdgcn_wmma_f32_16x16x32_f16(
                    false, a1.v, false, bf[ni].v, (short)0, acc[1][ni], false, false);
            }
        }
    }

    // ---------------- epilogue -------------------------------------------
    #pragma unroll
    for (int mi = 0; mi < 2; ++mi) {
        #pragma unroll
        for (int ni = 0; ni < 4; ++ni) {
            V8F u; u.v = acc[mi][ni];
            int col = colBase + wn * 64 + ni * 16 + ncol;
            if (col >= N) continue;
            float bv = bias ? bias[col] : 0.0f;
            #pragma unroll
            for (int r = 0; r < 8; ++r) {
                int row = rowBase + wm * 32 + mi * 16 + half * 8 + r;
                if (row >= M) continue;
                float v = u.f[r] + bv;
                if (act == 1) v = fmaxf(v, 0.0f);
                C[(size_t)row * N + col] = v;
            }
        }
    }
}

// ---------------------------------------------------------------------------
// Aux kernels
// ---------------------------------------------------------------------------
__global__ void zero_kernel(float* __restrict__ p, size_t n) {
    size_t i = (size_t)blockIdx.x * blockDim.x + threadIdx.x;
    if (i < n) p[i] = 0.0f;
}

__global__ void degree_kernel(const int* __restrict__ dst, float* __restrict__ cnt, int E) {
    int e = blockIdx.x * blockDim.x + threadIdx.x;
    if (e < E) atomicAdd(&cnt[dst[e]], 1.0f);
}

__global__ void invdeg_kernel(const float* __restrict__ cnt, float* __restrict__ inv, int n) {
    int i = blockIdx.x * blockDim.x + threadIdx.x;
    if (i < n) inv[i] = 1.0f / fmaxf(cnt[i], 1.0f);
}

// One block per edge: broadcast src/dst, coalesced row atomic-add.
__global__ void scatter_add_kernel(const float* __restrict__ x,
                                   const int* __restrict__ src,
                                   const int* __restrict__ dst,
                                   float* __restrict__ msum, int F) {
    int e = blockIdx.x;
    int s = src[e], d = dst[e];
    const float* xr = x    + (size_t)s * F;
    float*       mr = msum + (size_t)d * F;
    for (int f = threadIdx.x; f < F; f += blockDim.x)
        atomicAdd(&mr[f], xr[f]);
}

// Segment-max over post-ReLU values (>=0): uint bit-pattern atomicMax is exact.
__global__ void segmax_kernel(const float* __restrict__ h,
                              const int* __restrict__ batch,
                              float* __restrict__ g, size_t total, int F) {
    size_t i = (size_t)blockIdx.x * blockDim.x + threadIdx.x;
    if (i >= total) return;
    int node = (int)(i / F);
    int f    = (int)(i % F);
    atomicMax((unsigned int*)&g[(size_t)batch[node] * F + f],
              __float_as_uint(h[i]));
}

// conv1d(K=8) + ReLU + maxpool3 fused; relu(max)=max(relu) since relu monotone.
__global__ void conv1_kernel(const float* __restrict__ x, const float* __restrict__ w,
                             const float* __restrict__ b, float* __restrict__ out) {
    const int LO = 4375;
    int idx = blockIdx.x * blockDim.x + threadIdx.x;
    if (idx >= NB * 32 * LO) return;
    int opos = idx % LO;
    int c    = (idx / LO) & 31;
    int bt   = idx / (LO * 32);
    const float* in = x + (size_t)bt * LMUT + 3 * opos;
    float s0 = b[c], s1 = s0, s2 = s0;
    float xv[10];
    #pragma unroll
    for (int t = 0; t < 10; ++t) xv[t] = in[t];
    #pragma unroll
    for (int k = 0; k < 8; ++k) {
        float wv = w[c * 8 + k];
        s0 += xv[k] * wv; s1 += xv[k + 1] * wv; s2 += xv[k + 2] * wv;
    }
    out[idx] = fmaxf(fmaxf(fmaxf(s0, s1), s2), 0.0f);
}

__global__ void conv2_kernel(const float* __restrict__ x, const float* __restrict__ w,
                             const float* __restrict__ b, float* __restrict__ out) {
    const int LI = 4375, LO = 1456, CI = 32, CO = 64;
    int idx = blockIdx.x * blockDim.x + threadIdx.x;
    if (idx >= NB * CO * LO) return;
    int opos = idx % LO;
    int co   = (idx / LO) & (CO - 1);
    int bt   = idx / (LO * CO);
    const float* inb = x + (size_t)bt * CI * LI + 3 * opos;
    const float* wb  = w + (size_t)co * CI * 8;
    float s0 = b[co], s1 = s0, s2 = s0;
    for (int ci = 0; ci < CI; ++ci) {
        const float* ir = inb + (size_t)ci * LI;
        const float* wk = wb + ci * 8;
        float xv[10];
        #pragma unroll
        for (int t = 0; t < 10; ++t) xv[t] = ir[t];
        #pragma unroll
        for (int k = 0; k < 8; ++k) {
            float wv = wk[k];
            s0 += xv[k] * wv; s1 += xv[k + 1] * wv; s2 += xv[k + 2] * wv;
        }
    }
    out[idx] = fmaxf(fmaxf(fmaxf(s0, s1), s2), 0.0f);
}

__global__ void conv3_kernel(const float* __restrict__ x, const float* __restrict__ w,
                             const float* __restrict__ b, float* __restrict__ out) {
    const int LI = 1456, LO = 483, CI = 64, CO = 128;
    int idx = blockIdx.x * blockDim.x + threadIdx.x;
    if (idx >= NB * CO * LO) return;
    int opos = idx % LO;
    int co   = (idx / LO) & (CO - 1);
    int bt   = idx / (LO * CO);
    const float* inb = x + (size_t)bt * CI * LI + 3 * opos;
    const float* wb  = w + (size_t)co * CI * 8;
    float s0 = b[co], s1 = s0, s2 = s0;
    for (int ci = 0; ci < CI; ++ci) {
        const float* ir = inb + (size_t)ci * LI;
        const float* wk = wb + ci * 8;
        float xv[10];
        #pragma unroll
        for (int t = 0; t < 10; ++t) xv[t] = ir[t];
        #pragma unroll
        for (int k = 0; k < 8; ++k) {
            float wv = wk[k];
            s0 += xv[k] * wv; s1 += xv[k + 1] * wv; s2 += xv[k + 2] * wv;
        }
    }
    out[idx] = fmaxf(fmaxf(fmaxf(s0, s1), s2), 0.0f);
}

__global__ void head_out_kernel(const float* __restrict__ f2, const float* __restrict__ Wout,
                                const float* __restrict__ bout, float* __restrict__ out) {
    int bt = blockIdx.x * blockDim.x + threadIdx.x;
    if (bt >= NB) return;
    float s = bout[0];
    #pragma unroll 8
    for (int k = 0; k < 128; ++k) s += f2[bt * 128 + k] * Wout[k];
    out[bt] = 1.0f / (1.0f + expf(-s));
}

// ---------------------------------------------------------------------------
// Host launch
// ---------------------------------------------------------------------------
static inline void launch_gemm(const float* A, const float* B, const float* A2,
                               const float* B2, const float* scl, const float* bias,
                               float* C, int M, int N, int K, int act, hipStream_t s) {
    dim3 grid((N + BN - 1) / BN, (M + BM - 1) / BM);
    if (A2) {
        if (scl) gemm_kernel<true, true ><<<grid, 256, 0, s>>>(A, B, A2, B2, scl, bias, C, M, N, K, act);
        else     gemm_kernel<true, false><<<grid, 256, 0, s>>>(A, B, A2, B2, scl, bias, C, M, N, K, act);
    } else {
        if (scl) gemm_kernel<false, true ><<<grid, 256, 0, s>>>(A, B, A2, B2, scl, bias, C, M, N, K, act);
        else     gemm_kernel<false, false><<<grid, 256, 0, s>>>(A, B, A2, B2, scl, bias, C, M, N, K, act);
    }
}
static inline void launch_zero(float* p, size_t n, hipStream_t s) {
    zero_kernel<<<(unsigned)((n + 255) / 256), 256, 0, s>>>(p, n);
}

extern "C" void kernel_launch(void* const* d_in, const int* in_sizes, int n_in,
                              void* d_out, int out_size, void* d_ws, size_t ws_size,
                              hipStream_t stream) {
    const float* x      = (const float*)d_in[0];
    const int*   ei     = (const int*)d_in[1];
    const int*   batch  = (const int*)d_in[2];
    const float* xmut   = (const float*)d_in[3];
    // d_in[4] = edge_feat (unused by reference)
    const float *W1l = (const float*)d_in[5],  *b1l = (const float*)d_in[6],  *W1r = (const float*)d_in[7];
    const float *W2l = (const float*)d_in[8],  *b2l = (const float*)d_in[9],  *W2r = (const float*)d_in[10];
    const float *W3l = (const float*)d_in[11], *b3l = (const float*)d_in[12], *W3r = (const float*)d_in[13];
    const float *Wg1 = (const float*)d_in[14], *bg1 = (const float*)d_in[15];
    const float *Wg2 = (const float*)d_in[16], *bg2 = (const float*)d_in[17];
    const float *Wc1 = (const float*)d_in[18], *bc1 = (const float*)d_in[19];
    const float *Wc2 = (const float*)d_in[20], *bc2 = (const float*)d_in[21];
    const float *Wc3 = (const float*)d_in[22], *bc3 = (const float*)d_in[23];
    const float *Wxt = (const float*)d_in[24], *bxt = (const float*)d_in[25];
    const float *Wf1 = (const float*)d_in[26], *bf1 = (const float*)d_in[27];
    const float *Wf2 = (const float*)d_in[28], *bf2 = (const float*)d_in[29];
    const float *Wout= (const float*)d_in[30], *bout= (const float*)d_in[31];
    float* out = (float*)d_out;

    const int E = in_sizes[1] / 2;
    const int* src = ei;
    const int* dst = ei + E;

    // ---- workspace arena (static offsets; conv buffers alias dead graph bufs)
    char* w = (char*)d_ws;
    size_t off = 0;
    auto arena = [&](size_t bytes) -> float* {
        float* p = (float*)(w + off);
        off += (bytes + 255) & ~(size_t)255;
        return p;
    };
    float* cnt  = arena((size_t)NN * 4);
    float* invd = arena((size_t)NN * 4);
    float* msum = arena((size_t)NN * F2 * 4);        // 267 MB (max F)
    float* h1   = arena((size_t)NN * F1 * 4);        // 134 MB
    float* h2   = arena((size_t)NN * F2 * 4);        // 267 MB
    float* h3   = arena((size_t)NN * F3 * 4);        // 534 MB
    float* g    = arena((size_t)NB * F3 * 4);
    float* g1   = arena((size_t)NB * 1024 * 4);
    float* g2   = arena((size_t)NB * 128 * 4);
    float* xt   = arena((size_t)NB * 128 * 4);
    float* f1   = arena((size_t)NB * 1024 * 4);
    float* f2   = arena((size_t)NB * 128 * 4);
    float* c1 = h3;   // 143 MB <= 534 MB (h3 dead by conv time)
    float* c2 = h2;   //  95 MB <= 267 MB
    float* c3 = h1;   //  63 MB <= 134 MB
    (void)ws_size; (void)n_in; (void)out_size;

    // ---- degree & reciprocal (reused by all 3 SAGE layers)
    launch_zero(cnt, NN, stream);
    degree_kernel<<<(E + 255) / 256, 256, 0, stream>>>(dst, cnt, E);
    invdeg_kernel<<<(NN + 255) / 256, 256, 0, stream>>>(cnt, invd, NN);

    // ---- SAGE layer 1: h1 = relu(mean_agg(x) @ W1l + b1l + x @ W1r)
    launch_zero(msum, (size_t)NN * F0, stream);
    scatter_add_kernel<<<E, 256, 0, stream>>>(x, src, dst, msum, F0);
    launch_gemm(msum, W1l, x, W1r, invd, b1l, h1, NN, F1, F0, 1, stream);

    // ---- SAGE layer 2
    launch_zero(msum, (size_t)NN * F1, stream);
    scatter_add_kernel<<<E, 256, 0, stream>>>(h1, src, dst, msum, F1);
    launch_gemm(msum, W2l, h1, W2r, invd, b2l, h2, NN, F2, F1, 1, stream);

    // ---- SAGE layer 3
    launch_zero(msum, (size_t)NN * F2, stream);
    scatter_add_kernel<<<E, 256, 0, stream>>>(h2, src, dst, msum, F2);
    launch_gemm(msum, W3l, h2, W3r, invd, b3l, h3, NN, F3, F2, 1, stream);

    // ---- global max pool over batch, then graph head
    launch_zero(g, (size_t)NB * F3, stream);
    {
        size_t total = (size_t)NN * F3;
        segmax_kernel<<<(unsigned)((total + 255) / 256), 256, 0, stream>>>(h3, batch, g, total, F3);
    }
    launch_gemm(g,  Wg1, nullptr, nullptr, nullptr, bg1, g1, NB, 1024, F3,  1, stream);
    launch_gemm(g1, Wg2, nullptr, nullptr, nullptr, bg2, g2, NB, 128, 1024, 0, stream);

    // ---- CNN branch (graph buffers dead -> aliased)
    conv1_kernel<<<(NB * 32 * 4375 + 255) / 256, 256, 0, stream>>>(xmut, Wc1, bc1, c1);
    conv2_kernel<<<(NB * 64 * 1456 + 255) / 256, 256, 0, stream>>>(c1, Wc2, bc2, c2);
    conv3_kernel<<<(NB * 128 * 483 + 255) / 256, 256, 0, stream>>>(c2, Wc3, bc3, c3);
    launch_gemm(c3, Wxt, nullptr, nullptr, nullptr, bxt, xt, NB, 128, 61824, 0, stream);

    // ---- head: concat([g2, xt]) @ Wf1 done as dual-pair GEMM (K=128 each)
    launch_gemm(g2, Wf1, xt, Wf1 + (size_t)128 * 1024, nullptr, bf1, f1, NB, 1024, 128, 1, stream);
    launch_gemm(f1, Wf2, nullptr, nullptr, nullptr, bf2, f2, NB, 128, 1024, 1, stream);
    head_out_kernel<<<1, 256, 0, stream>>>(f2, Wout, bout, out);
}
